// AutoRegressiveRNN_33363305955908
// MI455X (gfx1250) — matrix operand order repaired and checked by reading, not compiled, not run on hardware
//
#include <hip/hip_runtime.h>
#include <hip/hip_bf16.h>

// AutoRegressive LSTM decode on gfx1250 (MI455X).
// B=128, T=256, IN=512, H=2048, gates 4H=8192.
// bf16 WMMA (v_wmma_f32_16x16x32_bf16), fp32 accumulation, weights packed
// once into WMMA-native tile layouts and resident in the 192MB L2.
// Step GEMM: block = 4 M-tiles x 8 N-tiles; A tiles double-buffered through
// LDS via global_load_async_to_lds_b128 (+ s_wait_asynccnt), B fragments
// double-buffered in registers.

#define Bsz 128
#define Tn  256
#define INs 512
#define Hs  2048
#define G4  8192   // 4*H

typedef __bf16 bf16_t;
typedef __attribute__((ext_vector_type(16))) __bf16 v16bf;
typedef __attribute__((ext_vector_type(8)))  float  v8f;
typedef __attribute__((ext_vector_type(4)))  int    v4i;

#if defined(__gfx1250__) && \
    __has_builtin(__builtin_amdgcn_global_load_async_to_lds_b128) && \
    __has_builtin(__builtin_amdgcn_s_wait_asynccnt)
#define USE_ASYNC_LDS 1
#else
#define USE_ASYNC_LDS 0
#endif

// ---------- packed-layout index helpers (wave32 WMMA fragment layouts) ----
__device__ __forceinline__ int a_lane(int m, int kk) { return (kk & 8) ? (m + 16) : m; }
__device__ __forceinline__ int a_half(int kk)        { return (kk & 7) | ((kk & 16) >> 1); }
__device__ __forceinline__ int b_lane(int n, int kk) { return n + ((kk & 8) << 1); }
__device__ __forceinline__ int b_half(int kk)        { return (kk & 7) | ((kk & 16) >> 1); }

__device__ __forceinline__ float sigm(float x) { return 1.0f / (1.0f + __expf(-x)); }

// 16-byte stage of one A-tile granule into LDS (async DMA path on gfx1250)
__device__ __forceinline__ void stage16(const bf16_t* __restrict__ src, bf16_t* dst) {
#if USE_ASYNC_LDS
    __builtin_amdgcn_global_load_async_to_lds_b128(
        (__attribute__((address_space(1))) v4i*)(src),
        (__attribute__((address_space(3))) v4i*)(dst), 0, 0);
#else
    *(uint4*)dst = *(const uint4*)src;
#endif
}

__device__ __forceinline__ void pipe_barrier() {
#if USE_ASYNC_LDS
    __builtin_amdgcn_s_wait_asynccnt(0);
#endif
    __syncthreads();
}

// ---------- one-time weight packing (fp32 -> bf16, WMMA B-layout tiles) ----
__global__ void k_pack_weights(const float* __restrict__ Wih,
                               const float* __restrict__ Whh,
                               bf16_t* __restrict__ Wihp,
                               bf16_t* __restrict__ Whhp) {
    const long n_ih = (long)G4 * INs;
    const long n_hh = (long)G4 * Hs;
    long stride = (long)gridDim.x * blockDim.x;
    for (long i = (long)blockIdx.x * blockDim.x + threadIdx.x; i < n_ih + n_hh; i += stride) {
        if (i < n_ih) {
            int j = (int)(i / INs), k = (int)(i % INs);
            int nt = j >> 4, n = j & 15, kt = k >> 5, kk = k & 31;
            long d = ((long)(nt * (INs / 32) + kt)) * 512 + b_lane(n, kk) * 16 + b_half(kk);
            Wihp[d] = (bf16_t)Wih[i];
        } else {
            long ii = i - n_ih;
            int j = (int)(ii / Hs), k = (int)(ii % Hs);
            int nt = j >> 4, n = j & 15, kt = k >> 5, kk = k & 31;
            long d = ((long)(nt * (Hs / 32) + kt)) * 512 + b_lane(n, kk) * 16 + b_half(kk);
            Whhp[d] = (bf16_t)Whh[ii];
        }
    }
}

// ---------- one-time input packing (zero the autoregressive column) --------
__global__ void k_pack_input(const float* __restrict__ input,
                             const int* __restrict__ tgt,
                             bf16_t* __restrict__ Xp) {
    const long n = (long)Bsz * Tn * INs;
    const int target = *tgt;
    long stride = (long)gridDim.x * blockDim.x;
    for (long i = (long)blockIdx.x * blockDim.x + threadIdx.x; i < n; i += stride) {
        int b = (int)(i / ((long)Tn * INs));
        long rem = i % ((long)Tn * INs);
        int t = (int)(rem / INs), k = (int)(rem % INs);
        int r = t * Bsz + b;
        int mt = r >> 4, m = r & 15, kt = k >> 5, kk = k & 31;
        float v = (k == target) ? 0.0f : input[i];
        long d = ((long)(mt * (INs / 32) + kt)) * 512 + a_lane(m, kk) * 16 + a_half(kk);
        Xp[d] = (bf16_t)v;
    }
}

// ---------- per-call state init ------------------------------------------
__global__ void k_init_state(const float* __restrict__ h0,
                             const float* __restrict__ c0,
                             const float* __restrict__ b_ih,
                             const float* __restrict__ b_hh,
                             const float* __restrict__ Wih,
                             const float* __restrict__ first_input,
                             const int* __restrict__ tgt,
                             bf16_t* __restrict__ Hp,
                             float* __restrict__ c,
                             float* __restrict__ bias,
                             float* __restrict__ wihc,
                             float* __restrict__ prev) {
    const int target = *tgt;
    const long n = (long)Bsz * Hs;
    long stride = (long)gridDim.x * blockDim.x;
    for (long i = (long)blockIdx.x * blockDim.x + threadIdx.x; i < n; i += stride) {
        int b = (int)(i / Hs), j = (int)(i % Hs);
        c[i] = c0[i];
        int mt = b >> 4, m = b & 15, kt = j >> 5, kk = j & 31;
        long d = ((long)(mt * (Hs / 32) + kt)) * 512 + a_lane(m, kk) * 16 + a_half(kk);
        Hp[d] = (bf16_t)h0[i];
        if (i < G4) {
            bias[i] = b_ih[i] + b_hh[i];
            wihc[i] = Wih[i * (long)INs + target];
        }
        if (i < Bsz) prev[i] = first_input[i];
    }
}

// ---------- per-step GEMM: gates = bias + x_t @ Wih^T + h @ Whh^T ----------
// Grid 128 blocks: mtg = blk>>6 (2 groups of 4 M-tiles), 8 N-tiles/block
// (one per wave). Wave holds 4 accumulators; A via LDS double buffer,
// B fragment double-buffered in registers.
__global__ void __launch_bounds__(256)
k_step_gemm(const bf16_t* __restrict__ Xp,
            const bf16_t* __restrict__ Wihp,
            const bf16_t* __restrict__ Whhp,
            const bf16_t* __restrict__ Hp,
            const float* __restrict__ bias,
            float* __restrict__ gates,
            int t) {
    __shared__ bf16_t Asmem[2][4][512];          // 2 x (4 tiles x 1KB)

    const int wid  = threadIdx.x >> 5;
    const int lane = threadIdx.x & 31;
    const int mtg  = blockIdx.x >> 6;            // 0..1 -> M-tile base mtg*4
    const int nt   = (blockIdx.x & 63) * 8 + wid; // 0..511

    // staging role: 64 threads per tile, 16B per thread
    const int smi  = threadIdx.x >> 6;           // which of the 4 A-tiles
    const int soff = (threadIdx.x & 63) * 8;     // element offset (16B granule)

    const bf16_t* A1 = Xp + ((long)(t * 8 + mtg * 4) * (INs / 32)) * 512; // +(mi*16+kt)*512
    const bf16_t* A2 = Hp + ((long)(mtg * 4) * (Hs / 32)) * 512;          // +(mi*64+kt)*512
    const bf16_t* B1 = Wihp + (long)nt * (INs / 32) * 512 + lane * 16;    // +kt*512
    const bf16_t* B2 = Whhp + (long)nt * (Hs / 32) * 512 + lane * 16;     // +kt*512

    v8f acc[4];
    {
        float bv = bias[nt * 16 + (lane & 15)];
#pragma unroll
        for (int mi = 0; mi < 4; ++mi)
#pragma unroll
            for (int v = 0; v < 8; ++v) acc[mi][v] = bv;
    }

    // prologue: stage phase-1 kt=0 into buffer 0
    stage16(A1 + ((long)smi * 16) * 512 + soff, &Asmem[0][smi][soff]);
    pipe_barrier();

    v16bf bfrag = *(const v16bf*)(B1);

    // ---- phase 1: K over IN=512 (16 k-tiles) ----
    for (int kt = 0; kt < 16; ++kt) {
        const int p = kt & 1;
        if (kt < 15) stage16(A1 + ((long)smi * 16 + kt + 1) * 512 + soff, &Asmem[p ^ 1][smi][soff]);
        else         stage16(A2 + ((long)smi * 64) * 512 + soff,          &Asmem[p ^ 1][smi][soff]);
        v16bf bnext = (kt < 15) ? *(const v16bf*)(B1 + (long)(kt + 1) * 512)
                                : *(const v16bf*)(B2);
#pragma unroll
        for (int mi = 0; mi < 4; ++mi) {
            v16bf a = *(const v16bf*)&Asmem[p][mi][lane * 16];
            acc[mi] = __builtin_amdgcn_wmma_f32_16x16x32_bf16(false, a, false, bfrag,
                                                              (short)0, acc[mi], false, false);
        }
        bfrag = bnext;
        pipe_barrier();
    }

    // ---- phase 2: K over H=2048 (64 k-tiles), weights resident in L2 ----
    for (int kt = 0; kt < 64; ++kt) {
        const int p = kt & 1;
        if (kt < 63) stage16(A2 + ((long)smi * 64 + kt + 1) * 512 + soff, &Asmem[p ^ 1][smi][soff]);
        if (kt + 8 < 64) __builtin_prefetch(B2 + (long)(kt + 8) * 512, 0, 0);
        v16bf bnext = (kt < 63) ? *(const v16bf*)(B2 + (long)(kt + 1) * 512) : bfrag;
#pragma unroll
        for (int mi = 0; mi < 4; ++mi) {
            v16bf a = *(const v16bf*)&Asmem[p][mi][lane * 16];
            acc[mi] = __builtin_amdgcn_wmma_f32_16x16x32_bf16(false, a, false, bfrag,
                                                              (short)0, acc[mi], false, false);
        }
        bfrag = bnext;
        if (kt < 63) pipe_barrier();
    }

    // C/D layout: VGPR v, lanes 0-15 -> M=v, N=lane; lanes 16-31 -> M=v+8.
    const int rhalf = (lane >> 4) << 3;
    const int col   = lane & 15;
#pragma unroll
    for (int mi = 0; mi < 4; ++mi) {
        const int rowb = (mtg * 4 + mi) * 16 + rhalf;
        const long cb  = (long)nt * 16 + col;
#pragma unroll
        for (int v = 0; v < 8; ++v)
            gates[(long)(rowb + v) * G4 + cb] = acc[mi][v];
    }
}

// ---------- per-step pointwise: LSTM cell + projection + feedback ----------
__global__ void __launch_bounds__(256)
k_step_pointwise(const float* __restrict__ gates,
                 const float* __restrict__ wihc,
                 float* __restrict__ c,
                 bf16_t* __restrict__ Hp,
                 const float* __restrict__ Wout,
                 const float* __restrict__ b_out,
                 float* __restrict__ prev,
                 float* __restrict__ out,
                 int t) {
    const int b  = blockIdx.x;
    const float pv = prev[b];
    float acc = 0.0f;
#pragma unroll
    for (int r = 0; r < Hs / 256; ++r) {
        const int j = threadIdx.x + (r << 8);
        const long gi = (long)b * G4 + j;
        const float xi = gates[gi]          + pv * wihc[j];
        const float xf = gates[gi + Hs]     + pv * wihc[j + Hs];
        const float xg = gates[gi + 2 * Hs] + pv * wihc[j + 2 * Hs];
        const float xo = gates[gi + 3 * Hs] + pv * wihc[j + 3 * Hs];
        const float cp = c[(long)b * Hs + j];
        const float cn = sigm(xf) * cp + sigm(xi) * tanhf(xg);
        const float h  = sigm(xo) * tanhf(cn);
        c[(long)b * Hs + j] = cn;
        const int mt = b >> 4, m = b & 15, kt = j >> 5, kk = j & 31;
        Hp[((long)(mt * (Hs / 32) + kt)) * 512 + a_lane(m, kk) * 16 + a_half(kk)] = (bf16_t)h;
        acc += h * Wout[j];
    }
    __shared__ float red[256];
    red[threadIdx.x] = acc;
    __syncthreads();
#pragma unroll
    for (int s = 128; s > 0; s >>= 1) {
        if (threadIdx.x < s) red[threadIdx.x] += red[threadIdx.x + s];
        __syncthreads();
    }
    if (threadIdx.x == 0) {
        const float y = red[0] + b_out[0];
        out[(long)b * Tn + t] = y;   // output shape (B, T, 1)
        prev[b] = y;                 // autoregressive feedback
    }
}

extern "C" void kernel_launch(void* const* d_in, const int* in_sizes, int n_in,
                              void* d_out, int out_size, void* d_ws, size_t ws_size,
                              hipStream_t stream) {
    const float* input       = (const float*)d_in[0];
    const float* first_input = (const float*)d_in[1];
    const float* h0          = (const float*)d_in[2];
    const float* c0          = (const float*)d_in[3];
    const float* W_ih        = (const float*)d_in[4];
    const float* W_hh        = (const float*)d_in[5];
    const float* b_ih        = (const float*)d_in[6];
    const float* b_hh        = (const float*)d_in[7];
    const float* W_out       = (const float*)d_in[8];
    const float* b_out       = (const float*)d_in[9];
    const int*   tgt         = (const int*)d_in[10];
    float* out = (float*)d_out;

    char* p = (char*)d_ws;
    bf16_t* Whhp = (bf16_t*)p; p += (size_t)G4 * Hs * sizeof(bf16_t);        // 32 MB
    bf16_t* Wihp = (bf16_t*)p; p += (size_t)G4 * INs * sizeof(bf16_t);       //  8 MB
    bf16_t* Xp   = (bf16_t*)p; p += (size_t)Bsz * Tn * INs * sizeof(bf16_t); // 32 MB
    bf16_t* Hp   = (bf16_t*)p; p += (size_t)Bsz * Hs * sizeof(bf16_t);       // 0.5 MB
    float*  cbuf = (float*)p;  p += (size_t)Bsz * Hs * sizeof(float);        //  1 MB
    float*  gbuf = (float*)p;  p += (size_t)Bsz * G4 * sizeof(float);        //  4 MB
    float*  bias = (float*)p;  p += (size_t)G4 * sizeof(float);
    float*  wihc = (float*)p;  p += (size_t)G4 * sizeof(float);
    float*  prev = (float*)p;  p += 256;

    k_pack_weights<<<4096, 256, 0, stream>>>(W_ih, W_hh, Wihp, Whhp);
    k_pack_input  <<<4096, 256, 0, stream>>>(input, tgt, Xp);
    k_init_state  <<<1024, 256, 0, stream>>>(h0, c0, b_ih, b_hh, W_ih, first_input,
                                             tgt, Hp, cbuf, bias, wihc, prev);

    for (int t = 0; t < Tn; ++t) {
        k_step_gemm     <<<128, 256, 0, stream>>>(Xp, Wihp, Whhp, Hp, bias, gbuf, t);
        k_step_pointwise<<<128, 256, 0, stream>>>(gbuf, wihc, cbuf, Hp, W_out, b_out,
                                                  prev, out, t);
    }
}
